// GCN_6597069766679
// MI455X (gfx1250) — matrix operand order repaired and checked by reading, not compiled, not run on hardware
//
#include <hip/hip_runtime.h>
#include <hip/hip_bf16.h>
#include <stdint.h>

#define NN 16384   // nodes
#define NF 512     // input features
#define NH 128     // hidden
#define NC 16      // classes

typedef __attribute__((ext_vector_type(16))) __bf16 v16bf;
typedef __attribute__((ext_vector_type(8)))  float  v8f;
typedef __attribute__((ext_vector_type(2)))  float  v2f;
typedef __attribute__((ext_vector_type(4)))  float  fv4;

struct __align__(16) U4 { unsigned int x, y, z, w; };
struct __align__(16) F4 { float x, y, z, w; };

union FragBF { v16bf v; U4 q[2]; };

#if defined(__has_builtin)
#if __has_builtin(__builtin_amdgcn_wmma_f32_16x16x4_f32)
#define HAVE_WMMA_F32X4 1
#endif
#if __has_builtin(__builtin_amdgcn_global_load_async_to_lds_b128) && \
    __has_builtin(__builtin_amdgcn_s_wait_asynccnt)
#define HAVE_ASYNC_LDS 1
#endif
#endif
#ifndef HAVE_WMMA_F32X4
#define HAVE_WMMA_F32X4 0
#endif
#ifndef HAVE_ASYNC_LDS
#define HAVE_ASYNC_LDS 0
#endif

// native RNE f32->bf16 (lets the compiler emit v_cvt_pk_bf16_f32 for pairs)
__device__ __forceinline__ unsigned short bf16u(float f) {
  return __builtin_bit_cast(unsigned short, (__bf16)f);
}
__device__ __forceinline__ unsigned int cvt2bf(float lo, float hi) {
  return (unsigned int)bf16u(lo) | ((unsigned int)bf16u(hi) << 16);
}

#if HAVE_ASYNC_LDS
// Builtin signature (from hipcc diagnostic): param0 is a pointer to
// __vector_size__(16) int in the global addrspace (printed as __device__),
// param1 the LDS-side equivalent. Hop address spaces via integer casts:
// generic->AS1 keeps the 64-bit global VA; generic->AS3 keeps addr[31:0],
// which is the LDS offset per the flat-aperture mapping.
typedef int b128i __attribute__((vector_size(4 * sizeof(int))));
__device__ __forceinline__ void async_b128(const void* g, const void* l) {
  __builtin_amdgcn_global_load_async_to_lds_b128(
      (__attribute__((address_space(1))) b128i*)(unsigned long long)g,
      (__attribute__((address_space(3))) b128i*)(unsigned int)(unsigned long long)l,
      0, 0);
}
#endif

// ---------------------------------------------------------------------------
// k1: XWt[c][i] = bf16( (x @ W1 + b1)[i][c] ), f32 WMMA (16x16x4) for accuracy.
// Block: 64 rows x 128 cols, 8 waves (rt = w&3 row-tile, cg = (w>>2)*64 cols).
// ---------------------------------------------------------------------------
__global__ __launch_bounds__(256) void gcn_k1_xw1(const float* __restrict__ x,
                                                  const float* __restrict__ W1,
                                                  const float* __restrict__ b1,
                                                  unsigned short* __restrict__ xwt) {
  __shared__ __align__(16) float lX[64 * 16];    // [row][k]   4 KB
  __shared__ __align__(16) float lW[128 * 16];   // [col][k]   8 KB (transposed)
  const int t = threadIdx.x;
  const int wave = t >> 5, lane = t & 31;
  const int m = lane & 15, half = lane >> 4;
  const int rt = wave & 3, cg = (wave >> 2) * 64;
  const int r0 = blockIdx.x * 64;

  v8f acc[4] = {};

  for (int k0 = 0; k0 < NF; k0 += 16) {
    {
      const int r = t >> 2, q = t & 3;
      *(F4*)(lX + r * 16 + q * 4) =
          *(const F4*)(x + (size_t)(r0 + r) * NF + k0 + q * 4);
#pragma unroll
      for (int j = 0; j < 8; ++j) {
        const int e = t * 8 + j;              // 2048 elements
        const int kk = e >> 7, c = e & 127;
        lW[c * 16 + kk] = W1[(size_t)(k0 + kk) * NH + c];
      }
    }
    __syncthreads();
#if HAVE_WMMA_F32X4
#pragma unroll
    for (int kk = 0; kk < 16; kk += 4) {
      const float* ap = lX + (rt * 16 + m) * 16 + kk + 2 * half;
      v2f a = { ap[0], ap[1] };
#pragma unroll
      for (int ct = 0; ct < 4; ++ct) {
        const float* bp = lW + (cg + ct * 16 + m) * 16 + kk + 2 * half;
        v2f b = { bp[0], bp[1] };
        acc[ct] = __builtin_amdgcn_wmma_f32_16x16x4_f32(
            false, a, false, b, (short)0, acc[ct], false, false);
      }
    }
#else
#pragma unroll
    for (int kk = 0; kk < 16; ++kk) {
      float xv[8];
#pragma unroll
      for (int v = 0; v < 8; ++v)
        xv[v] = lX[(rt * 16 + 8 * half + v) * 16 + kk];
#pragma unroll
      for (int ct = 0; ct < 4; ++ct) {
        const float wv = lW[(cg + ct * 16 + m) * 16 + kk];
#pragma unroll
        for (int v = 0; v < 8; ++v) acc[ct][v] = fmaf(xv[v], wv, acc[ct][v]);
      }
    }
#endif
    __syncthreads();
  }
  // C/D layout: lane holds col n=m, rows rt*16 + 8*half + (0..7) -> one b128
  // transposed store per tile.
#pragma unroll
  for (int ct = 0; ct < 4; ++ct) {
    const int col = cg + ct * 16 + m;
    const float bias = b1[col];
    U4 u = { cvt2bf(acc[ct][0] + bias, acc[ct][1] + bias),
             cvt2bf(acc[ct][2] + bias, acc[ct][3] + bias),
             cvt2bf(acc[ct][4] + bias, acc[ct][5] + bias),
             cvt2bf(acc[ct][6] + bias, acc[ct][7] + bias) };
    *(U4*)(xwt + (size_t)col * NN + r0 + rt * 16 + 8 * half) = u;
  }
}

// ---------------------------------------------------------------------------
// k2: H = adj @ XW (bf16 WMMA, f32 acc), fused ReLU and G = relu(H)@W2 + b2.
// Block: 64 rows x 128 cols. adj streamed non-temporally, cvt to bf16 on fly.
// XWt tile staged global->LDS via async-to-LDS when available (ASYNCcnt).
// ---------------------------------------------------------------------------
__global__ __launch_bounds__(256) void gcn_k2_aggr(const float* __restrict__ adj,
                                                   const unsigned short* __restrict__ xwt,
                                                   const float* __restrict__ W2,
                                                   const float* __restrict__ b2,
                                                   unsigned short* __restrict__ gt) {
  __shared__ __align__(16) unsigned short lA[64 * 32];    //  4 KB  [row][k] bf16
  __shared__ __align__(16) unsigned short lB[128 * 32];   //  8 KB  [col][k] bf16
  __shared__ __align__(16) unsigned short lH[64 * 128];   // 16 KB  relu(H) tile
  __shared__ __align__(16) unsigned short lW2[16 * 128];  //  4 KB  [n][k] bf16
  const int t = threadIdx.x;
  const int wave = t >> 5, lane = t & 31;
  const int m = lane & 15, half = lane >> 4;
  const int rt = wave & 3, cg = (wave >> 2) * 64;
  const int r0 = blockIdx.x * 64;

  // stage W2 transposed bf16 once (e = k*16+n, coalesced read)
#pragma unroll
  for (int j = 0; j < 8; ++j) {
    const int e = t * 8 + j;
    const int kk = e >> 4, n = e & 15;
    lW2[n * 128 + kk] = bf16u(W2[e]);
  }

  const int ar = t >> 2, aq = t & 3;   // adj: row 0..63, 8-col quad
  const int bc = t >> 1, bh = t & 1;   // xwt: col 0..127, 16-k half
  const float* aprow = adj + (size_t)(r0 + ar) * NN + aq * 8;
  const unsigned short* bprow = xwt + (size_t)bc * NN + bh * 16;
  unsigned short* lbdst = lB + bc * 32 + bh * 16;

  fv4 pa0 = __builtin_nontemporal_load((const fv4*)aprow);
  fv4 pa1 = __builtin_nontemporal_load((const fv4*)(aprow + 4));
#if HAVE_ASYNC_LDS
  async_b128(bprow, lbdst);
  async_b128(bprow + 8, lbdst + 8);
#else
  U4 pb0 = *(const U4*)bprow;
  U4 pb1 = *(const U4*)(bprow + 8);
#endif

  v8f acc[4] = {};

  for (int k0 = 0; k0 < NN; k0 += 32) {
    {
      U4 u = { cvt2bf(pa0[0], pa0[1]), cvt2bf(pa0[2], pa0[3]),
               cvt2bf(pa1[0], pa1[1]), cvt2bf(pa1[2], pa1[3]) };
      *(U4*)(lA + ar * 32 + aq * 8) = u;
#if HAVE_ASYNC_LDS
      __builtin_amdgcn_s_wait_asynccnt(0);   // this thread's lB chunks arrived
#else
      *(U4*)lbdst = pb0;
      *(U4*)(lbdst + 8) = pb1;
#endif
    }
    __syncthreads();
    if (k0 + 32 < NN) {  // register prefetch of next adj slab under the WMMAs
      pa0 = __builtin_nontemporal_load((const fv4*)(aprow + k0 + 32));
      pa1 = __builtin_nontemporal_load((const fv4*)(aprow + k0 + 36));
#if !HAVE_ASYNC_LDS
      pb0 = *(const U4*)(bprow + k0 + 32);
      pb1 = *(const U4*)(bprow + k0 + 40);
#endif
    }
    FragBF a;  // 16-bit A 16x32: e0..7 = K 8h+0..7, e8..15 = K 16+8h+0..7
    const unsigned short* ap = lA + (rt * 16 + m) * 32;
    a.q[0] = *(const U4*)(ap + 8 * half);
    a.q[1] = *(const U4*)(ap + 16 + 8 * half);
#pragma unroll
    for (int ct = 0; ct < 4; ++ct) {
      FragBF b;  // 16-bit B 32x16: lane n=m, e0..7 = K 8h+0..7, e8..15 = K 16+8h+0..7
      const unsigned short* bp = lB + (cg + ct * 16 + m) * 32 + 8 * half;
      b.q[0] = *(const U4*)bp;
      b.q[1] = *(const U4*)(bp + 16);
      acc[ct] = __builtin_amdgcn_wmma_f32_16x16x32_bf16(
          false, a.v, false, b.v, (short)0, acc[ct], false, false);
    }
    __syncthreads();
#if HAVE_ASYNC_LDS
    if (k0 + 32 < NN) {  // LDS free after the tail barrier: issue next slab
      async_b128(bprow + k0 + 32, lbdst);
      async_b128(bprow + k0 + 40, lbdst + 8);
    }
#endif
  }

  // ReLU -> lH (bf16 [64][128])
#pragma unroll
  for (int ct = 0; ct < 4; ++ct) {
#pragma unroll
    for (int v = 0; v < 8; ++v) {
      float h = acc[ct][v];
      h = h > 0.f ? h : 0.f;
      lH[(rt * 16 + 8 * half + v) * 128 + cg + ct * 16 + m] = bf16u(h);
    }
  }
  __syncthreads();

  // Fused: G = relu(H) @ W2 + b2, one 16x16 tile per wave (waves 0..3), K=128.
  if (wave < 4) {
    v8f g = {};
#pragma unroll
    for (int kk = 0; kk < 4; ++kk) {
      FragBF a2, bw;
      const unsigned short* ap = lH + (wave * 16 + m) * 128 + kk * 32;
      a2.q[0] = *(const U4*)(ap + 8 * half);
      a2.q[1] = *(const U4*)(ap + 16 + 8 * half);
      const unsigned short* bp = lW2 + m * 128 + kk * 32 + 8 * half;
      bw.q[0] = *(const U4*)bp;
      bw.q[1] = *(const U4*)(bp + 16);
      g = __builtin_amdgcn_wmma_f32_16x16x32_bf16(
          false, a2.v, false, bw.v, (short)0, g, false, false);
    }
    const float bias = b2[m];
    U4 u = { cvt2bf(g[0] + bias, g[1] + bias), cvt2bf(g[2] + bias, g[3] + bias),
             cvt2bf(g[4] + bias, g[5] + bias), cvt2bf(g[6] + bias, g[7] + bias) };
    *(U4*)(gt + (size_t)m * NN + r0 + wave * 16 + 8 * half) = u;  // Gt[n][row]
  }
}

// ---------------------------------------------------------------------------
// k3: out = adj @ G. Block: 128 rows x 16 cols, one 16x16 tile per wave.
// B fragments come straight from L2-resident Gt (row-contiguous in K).
// ---------------------------------------------------------------------------
__global__ __launch_bounds__(256) void gcn_k3_out(const float* __restrict__ adj,
                                                  const unsigned short* __restrict__ gt,
                                                  float* __restrict__ out) {
  __shared__ __align__(16) unsigned short lA[128 * 32];   // 8 KB
  const int t = threadIdx.x;
  const int wave = t >> 5, lane = t & 31;
  const int m = lane & 15, half = lane >> 4;
  const int r0 = blockIdx.x * 128;
  const int ar = t >> 1, ah = t & 1;   // row 0..127, 16-col half

  const float* aprow = adj + (size_t)(r0 + ar) * NN + ah * 16;
  fv4 p0 = __builtin_nontemporal_load((const fv4*)(aprow + 0));
  fv4 p1 = __builtin_nontemporal_load((const fv4*)(aprow + 4));
  fv4 p2 = __builtin_nontemporal_load((const fv4*)(aprow + 8));
  fv4 p3 = __builtin_nontemporal_load((const fv4*)(aprow + 12));

  v8f acc = {};

  for (int k0 = 0; k0 < NN; k0 += 32) {
    {
      U4 u0 = { cvt2bf(p0[0], p0[1]), cvt2bf(p0[2], p0[3]),
                cvt2bf(p1[0], p1[1]), cvt2bf(p1[2], p1[3]) };
      U4 u1 = { cvt2bf(p2[0], p2[1]), cvt2bf(p2[2], p2[3]),
                cvt2bf(p3[0], p3[1]), cvt2bf(p3[2], p3[3]) };
      *(U4*)(lA + ar * 32 + ah * 16) = u0;
      *(U4*)(lA + ar * 32 + ah * 16 + 8) = u1;
    }
    __syncthreads();
    if (k0 + 32 < NN) {
      const float* p = aprow + (k0 + 32);
      p0 = __builtin_nontemporal_load((const fv4*)(p + 0));
      p1 = __builtin_nontemporal_load((const fv4*)(p + 4));
      p2 = __builtin_nontemporal_load((const fv4*)(p + 8));
      p3 = __builtin_nontemporal_load((const fv4*)(p + 12));
    }
    FragBF a, b;
    const unsigned short* ap = lA + (wave * 16 + m) * 32;
    a.q[0] = *(const U4*)(ap + 8 * half);
    a.q[1] = *(const U4*)(ap + 16 + 8 * half);
    const unsigned short* bp = gt + (size_t)m * NN + k0 + 8 * half;
    b.q[0] = *(const U4*)(bp);
    b.q[1] = *(const U4*)(bp + 16);
    acc = __builtin_amdgcn_wmma_f32_16x16x32_bf16(
        false, a.v, false, b.v, (short)0, acc, false, false);
    __syncthreads();
  }
#pragma unroll
  for (int v = 0; v < 8; ++v)
    out[(size_t)(r0 + wave * 16 + 8 * half + v) * NC + m] = acc[v];
}

// ---------------------------------------------------------------------------
extern "C" void kernel_launch(void* const* d_in, const int* in_sizes, int n_in,
                              void* d_out, int out_size, void* d_ws, size_t ws_size,
                              hipStream_t stream) {
  (void)in_sizes; (void)n_in; (void)out_size; (void)ws_size;
  const float* x   = (const float*)d_in[0];
  const float* adj = (const float*)d_in[1];
  const float* W1  = (const float*)d_in[2];
  const float* b1  = (const float*)d_in[3];
  const float* W2  = (const float*)d_in[4];
  const float* b2  = (const float*)d_in[5];

  unsigned short* xwt = (unsigned short*)d_ws;            // [128][16384] bf16, 4 MB
  unsigned short* gt  = xwt + (size_t)NH * NN;            // [16][16384]  bf16, 0.5 MB

  gcn_k1_xw1<<<NN / 64, 256, 0, stream>>>(x, W1, b1, xwt);
  gcn_k2_aggr<<<NN / 64, 256, 0, stream>>>(adj, xwt, W2, b2, gt);
  gcn_k3_out<<<NN / 128, 256, 0, stream>>>(adj, gt, (float*)d_out);
}
